// GraphEncoder_51780125721127
// MI455X (gfx1250) — matrix operand order repaired and checked by reading, not compiled, not run on hardware
//
#include <hip/hip_runtime.h>
#include <hip/hip_bf16.h>
#include <stddef.h>

// ---------------------------------------------------------------------------
// GENConv GraphEncoder forward for MI455X (gfx1250, wave32, WMMA).
//   N=50000 nodes, E=800000 edges, G=64 graphs, IN=32, H=96, ED=16, OUT=128
// Dense GEMMs: split-precision f16 WMMA (A_hi*B_hi + A_hi*B_lo + A_lo*B_hi),
// one 16x32 output strip per wave (2 tiles), shapes templated so all strides
// are immediates (no spills, no runtime address math).
// Edge softmax-aggregation is fused (projection recomputed, 2 passes).
// ---------------------------------------------------------------------------

typedef __attribute__((ext_vector_type(16))) _Float16 v16h;
typedef __attribute__((ext_vector_type(8)))  float    v8f;

#define HDIM 96
#define EDIM 16
#define GENEPS 1e-7f
#define BNEPS  1e-5f

// ---------------------------------------------------------------------------
// Generic zero-fill
// ---------------------------------------------------------------------------
__global__ void k_zero(float* __restrict__ p, size_t n) {
    size_t i = (size_t)blockIdx.x * blockDim.x + threadIdx.x;
    if (i < n) p[i] = 0.0f;
}

// ---------------------------------------------------------------------------
// WMMA GEMM: C[M,Nc] = A[M,Kc] * B[Kc,Nc], fp32 storage, split-f16 multiply
// via v_wmma_f32_16x16x32_f16. Each wave: 16 rows x 32 cols. 4 waves/block.
// Requires M%16==0, Nc%32==0, Kc%32==0 (all compile-time here).
//
// Fragment layouts used here (wave32; A and B share contiguous k order):
//   A 16x32 f16 : lane l -> row m=l&15, half h=l>>4;
//                 elems 0..15 = K (k0 + 16h + 0..15)
//   B 32x16 f16 : lane l -> col n=l&15, half h=l>>4;
//                 elems 0..15 = K (k0 + 16h + 0..15)
//   C/D 16x16 f32: vgpr r, lane l -> row = r + 8h, col = l&15
// ---------------------------------------------------------------------------
template <int Nc, int Kc>
__global__ void k_gemm_wmma(const float* __restrict__ A,
                            const float* __restrict__ B,
                            float* __restrict__ C, int M) {
    constexpr int NSTRIPS = Nc / 32;
    const int mtiles = M >> 4;
    const int wave   = blockIdx.x * 4 + (threadIdx.x >> 5);
    if (wave >= mtiles * NSTRIPS) return;           // uniform per wave
    const int mt = wave / NSTRIPS;
    const int ns = wave % NSTRIPS;

    const int lane = threadIdx.x & 31;
    const int half = lane >> 4;
    const int r    = lane & 15;

    const float* __restrict__ Ar = A + (size_t)(mt * 16 + r) * Kc + half * 16;
    const float* __restrict__ Bc = B + (size_t)(half * 16) * Nc + ns * 32 + r;

    v8f acc0 = {}, acc1 = {};

#pragma unroll 1
    for (int k0 = 0; k0 < Kc; k0 += 32) {
        // ---- A fragment: one contiguous 64B run -> global_load_b128 ----
        const float4* __restrict__ Ap = (const float4*)(Ar + k0);
        float af[16];
        {
            float4 p0 = Ap[0], p1 = Ap[1], q0 = Ap[2], q1 = Ap[3]; // 16 contiguous floats
            af[0]=p0.x; af[1]=p0.y; af[2]=p0.z;  af[3]=p0.w;
            af[4]=p1.x; af[5]=p1.y; af[6]=p1.z;  af[7]=p1.w;
            af[8]=q0.x; af[9]=q0.y; af[10]=q0.z; af[11]=q0.w;
            af[12]=q1.x;af[13]=q1.y;af[14]=q1.z; af[15]=q1.w;
        }
        v16h ah, al;
#pragma unroll
        for (int i = 0; i < 16; ++i) {
            _Float16 hi = (_Float16)af[i];
            ah[i] = hi;
            al[i] = (_Float16)(af[i] - (float)hi);
        }

        const float* __restrict__ Bk = Bc + (size_t)k0 * Nc;

        // ---- tile 0 (cols ns*32 + 0..15) ----
        {
            v16h bh, bl;
#pragma unroll
            for (int i = 0; i < 16; ++i) {
                float bv = Bk[i * Nc];              // immediate offsets
                _Float16 hi = (_Float16)bv;
                bh[i] = hi;
                bl[i] = (_Float16)(bv - (float)hi);
            }
            acc0 = __builtin_amdgcn_wmma_f32_16x16x32_f16(
                false, ah, false, bh, (short)0, acc0, false, false);
            acc0 = __builtin_amdgcn_wmma_f32_16x16x32_f16(
                false, ah, false, bl, (short)0, acc0, false, false);
            acc0 = __builtin_amdgcn_wmma_f32_16x16x32_f16(
                false, al, false, bh, (short)0, acc0, false, false);
        }
        // ---- tile 1 (cols ns*32 + 16..31) ----
        {
            v16h bh, bl;
#pragma unroll
            for (int i = 0; i < 16; ++i) {
                float bv = Bk[i * Nc + 16];
                _Float16 hi = (_Float16)bv;
                bh[i] = hi;
                bl[i] = (_Float16)(bv - (float)hi);
            }
            acc1 = __builtin_amdgcn_wmma_f32_16x16x32_f16(
                false, ah, false, bh, (short)0, acc1, false, false);
            acc1 = __builtin_amdgcn_wmma_f32_16x16x32_f16(
                false, ah, false, bl, (short)0, acc1, false, false);
            acc1 = __builtin_amdgcn_wmma_f32_16x16x32_f16(
                false, al, false, bh, (short)0, acc1, false, false);
        }
    }

    // ---- store: rows mt*16 + 8*half + rr, immediate row offsets ----
    float* __restrict__ Crow = C + (size_t)(mt * 16 + 8 * half) * Nc
                                 + ns * 32 + r;
#pragma unroll
    for (int rr = 0; rr < 8; ++rr) {
        Crow[rr * Nc]      = acc0[rr];
        Crow[rr * Nc + 16] = acc1[rr];
    }
}

// ---------------------------------------------------------------------------
// Fused edge kernel.  blockDim=192 -> 2 edges/block, thread = (edge, channel).
//   msg[c] = relu(feat[src][c] + sum_k ea[e][k]*we[k][c]) + eps
// pass 0: atomic float-max (as int; messages are strictly positive)
// pass 1: den[dst][c] += exp(msg-m);  num[dst][c] += msg*exp(msg-m)
// ---------------------------------------------------------------------------
__global__ void k_edge_pass(const float* __restrict__ feat,
                            const float* __restrict__ ea,
                            const int*   __restrict__ eidx,
                            const float* __restrict__ wedge,
                            float* __restrict__ mmax,
                            float* __restrict__ num,
                            float* __restrict__ den,
                            int E, int pass) {
    __shared__ float sw[EDIM * HDIM];   // 16x96 projection weight
    __shared__ float sea[2 * EDIM];     // edge attrs for the 2 edges

    const int t = threadIdx.x;          // 0..191
    for (int i = t; i < EDIM * HDIM; i += 192) sw[i] = wedge[i];
    const long long e0 = (long long)blockIdx.x * 2;
    if (t < 2 * EDIM) {
        long long idx = e0 * EDIM + t;
        sea[t] = (idx < (long long)E * EDIM) ? ea[idx] : 0.0f;
    }
    __syncthreads();

    const int el = t / HDIM;            // 0 or 1
    const int c  = t - el * HDIM;       // 0..95
    const long long e = e0 + el;
    if (e >= E) return;

    const int src = eidx[e];
    const int dst = eidx[(size_t)E + e];

    const float* __restrict__ frow = feat + (size_t)src * HDIM;
    __builtin_prefetch(frow + c, 0, 3);             // global_prefetch_b8

    float p = 0.0f;
#pragma unroll
    for (int k = 0; k < EDIM; ++k) p += sea[el * EDIM + k] * sw[k * HDIM + c];

    float v   = frow[c] + p;
    float msg = (v > 0.0f ? v : 0.0f) + GENEPS;

    const size_t o = (size_t)dst * HDIM + c;
    if (pass == 0) {
        atomicMax(reinterpret_cast<int*>(mmax) + o, __float_as_int(msg));
    } else {
        float m  = mmax[o];
        float ex = __expf(msg - m);
        atomicAdd(den + o, ex);
        atomicAdd(num + o, msg * ex);
    }
}

// ---------------------------------------------------------------------------
// h = (den>0 ? num/den : 0) + dstfeat
// ---------------------------------------------------------------------------
__global__ void k_combine(const float* __restrict__ num,
                          const float* __restrict__ den,
                          const float* __restrict__ dstfeat,
                          float* __restrict__ h, size_t total) {
    size_t i = (size_t)blockIdx.x * blockDim.x + threadIdx.x;
    if (i >= total) return;
    float d = den[i];
    float a = (d > 0.0f) ? num[i] / d : 0.0f;
    h[i] = a + dstfeat[i];
}

// ---------------------------------------------------------------------------
// BatchNorm: column sum / sumsq (blockDim == C), then finalize, then apply.
// ---------------------------------------------------------------------------
__global__ void k_bn_stats(const float* __restrict__ x,
                           float* __restrict__ gsum, float* __restrict__ gsq,
                           int Nrows, int C) {
    const int c = threadIdx.x;
    float s = 0.0f, q = 0.0f;
    for (int r = blockIdx.x; r < Nrows; r += gridDim.x) {
        float v = x[(size_t)r * C + c];
        s += v; q += v * v;
    }
    atomicAdd(gsum + c, s);
    atomicAdd(gsq + c, q);
}

__global__ void k_bn_finalize(const float* __restrict__ gsum,
                              const float* __restrict__ gsq,
                              const float* __restrict__ g,
                              const float* __restrict__ b,
                              float* __restrict__ scale,
                              float* __restrict__ shift,
                              int Nrows, int C) {
    int c = blockIdx.x * blockDim.x + threadIdx.x;
    if (c >= C) return;
    float invn = 1.0f / (float)Nrows;
    float mean = gsum[c] * invn;
    float var  = gsq[c] * invn - mean * mean;
    float sc   = g[c] * rsqrtf(var + BNEPS);
    scale[c] = sc;
    shift[c] = b[c] - mean * sc;
}

__global__ void k_bn_relu(const float* __restrict__ x,
                          float* __restrict__ y,
                          const float* __restrict__ scale,
                          const float* __restrict__ shift,
                          size_t total, int C) {
    size_t i = (size_t)blockIdx.x * blockDim.x + threadIdx.x;
    if (i >= total) return;
    int c = (int)(i % C);
    float v = x[i] * scale[c] + shift[c];
    y[i] = v > 0.0f ? v : 0.0f;
}

// ---------------------------------------------------------------------------
// Global mean pool (G=64), then linear 96 -> 128.
// ---------------------------------------------------------------------------
__global__ void k_pool(const float* __restrict__ h,
                       const int* __restrict__ batch,
                       float* __restrict__ pooled, float* __restrict__ cnt,
                       int Nrows) {
    size_t i = (size_t)blockIdx.x * blockDim.x + threadIdx.x;
    if (i >= (size_t)Nrows * HDIM) return;
    int row = (int)(i / HDIM);
    int c   = (int)(i - (size_t)row * HDIM);
    int g   = batch[row];
    atomicAdd(pooled + (size_t)g * HDIM + c, h[i]);
    if (c == 0) atomicAdd(cnt + g, 1.0f);
}

__global__ void k_final_linear(const float* __restrict__ pooled,
                               const float* __restrict__ cnt,
                               const float* __restrict__ lw,  // 96x128
                               const float* __restrict__ lb,  // 128
                               float* __restrict__ out, int OUT) {
    const int g = blockIdx.x;     // 64
    const int o = threadIdx.x;    // 128
    float inv = 1.0f / fmaxf(cnt[g], 1.0f);
    float acc = lb[o];
#pragma unroll 8
    for (int c = 0; c < HDIM; ++c)
        acc += pooled[(size_t)g * HDIM + c] * inv * lw[(size_t)c * OUT + o];
    out[(size_t)g * OUT + o] = acc;
}

// ---------------------------------------------------------------------------
// Host-side orchestration
// ---------------------------------------------------------------------------
template <int Nc, int Kc>
static inline void launch_gemm(const float* A, const float* B, float* C,
                               int M, hipStream_t s) {
    int waves  = (M / 16) * (Nc / 32);
    int blocks = (waves + 3) / 4;
    k_gemm_wmma<Nc, Kc><<<blocks, 128, 0, s>>>(A, B, C, M);
}

extern "C" void kernel_launch(void* const* d_in, const int* in_sizes, int n_in,
                              void* d_out, int out_size, void* d_ws, size_t ws_size,
                              hipStream_t stream) {
    const float* x     = (const float*)d_in[0];
    const float* ea    = (const float*)d_in[1];
    const int*   eidx  = (const int*)  d_in[2];
    const int*   batch = (const int*)  d_in[3];

    const float* w_src1 = (const float*)d_in[4];
    const float* w_dst1 = (const float*)d_in[5];

    const float* w_edge[3] = { (const float*)d_in[6],  (const float*)d_in[13], (const float*)d_in[20] };
    const float* mlp_w1[3] = { (const float*)d_in[7],  (const float*)d_in[14], (const float*)d_in[21] };
    const float* mlp_g [3] = { (const float*)d_in[8],  (const float*)d_in[15], (const float*)d_in[22] };
    const float* mlp_b [3] = { (const float*)d_in[9],  (const float*)d_in[16], (const float*)d_in[23] };
    const float* mlp_w2[3] = { (const float*)d_in[10], (const float*)d_in[17], (const float*)d_in[24] };
    const float* bn_g  [3] = { (const float*)d_in[11], (const float*)d_in[18], (const float*)d_in[25] };
    const float* bn_b  [3] = { (const float*)d_in[12], (const float*)d_in[19], (const float*)d_in[26] };

    const float* lin_w = (const float*)d_in[27];
    const float* lin_b = (const float*)d_in[28];

    const int N  = in_sizes[0] / 32;     // 50000
    const int E  = in_sizes[1] / EDIM;   // 800000
    const int G  = 64;
    const int H = HDIM, H2 = 2 * HDIM, OUT = 128;

    // ---- workspace carve-up (floats) ----
    float* ws = (float*)d_ws;
    const size_t NH = (size_t)N * H;
    float* xsrc  = ws;                 // N*H  : layer-1 src feats / later layer output
    float* xdst  = xsrc  + NH;         // N*H  : layer-1 dst feats
    float* mmax  = xdst  + NH;         // N*H
    float* num   = mmax  + NH;         // N*H
    float* den   = num   + NH;         // N*H
    float* hbuf  = den   + NH;         // N*H
    float* ybuf  = hbuf  + NH;         // N*2H
    float* zbuf  = ybuf  + 2 * NH;     // N*2H
    float* gsum  = zbuf  + 2 * NH;     // 2H
    float* gsq   = gsum  + H2;         // 2H
    float* scale = gsq   + H2;         // 2H
    float* shift = scale + H2;         // 2H
    float* pooled= shift + H2;         // G*H
    float* cnt   = pooled + (size_t)G * H; // G
    (void)ws_size; (void)n_in;

    const int Z = 256;
    auto zero = [&](float* p, size_t n) {
        k_zero<<<(unsigned)((n + Z - 1) / Z), Z, 0, stream>>>(p, n);
    };

    // ---- layer-1 input projections (WMMA) ----
    launch_gemm<HDIM, 32>(x, w_src1, xsrc, N, stream);
    launch_gemm<HDIM, 32>(x, w_dst1, xdst, N, stream);

    const int eblocks = (E + 1) / 2;        // 2 edges per block, blockDim=192

    for (int l = 0; l < 3; ++l) {
        const float* srcfeat = xsrc;                 // layers 2,3 reuse xsrc output
        const float* dstfeat = (l == 0) ? xdst : xsrc;

        // softmax aggregation
        zero(mmax, 3 * NH);                          // mmax, num, den (contiguous)
        k_edge_pass<<<eblocks, 192, 0, stream>>>(srcfeat, ea, eidx, w_edge[l],
                                                 mmax, num, den, E, 0);
        k_edge_pass<<<eblocks, 192, 0, stream>>>(srcfeat, ea, eidx, w_edge[l],
                                                 mmax, num, den, E, 1);
        k_combine<<<(unsigned)((NH + Z - 1) / Z), Z, 0, stream>>>(num, den, dstfeat,
                                                                  hbuf, NH);

        // internal MLP: Lin(H,2H) -> BN -> ReLU -> Lin(2H,H)
        launch_gemm<2 * HDIM, HDIM>(hbuf, mlp_w1[l], ybuf, N, stream);
        zero(gsum, 2 * (size_t)H2);
        k_bn_stats<<<768, H2, 0, stream>>>(ybuf, gsum, gsq, N, H2);
        k_bn_finalize<<<1, H2, 0, stream>>>(gsum, gsq, mlp_g[l], mlp_b[l],
                                            scale, shift, N, H2);
        k_bn_relu<<<(unsigned)((2 * NH + Z - 1) / Z), Z, 0, stream>>>(ybuf, zbuf,
                                                                      scale, shift,
                                                                      2 * NH, H2);
        launch_gemm<HDIM, 2 * HDIM>(zbuf, mlp_w2[l], xsrc, N, stream); // overwrite xsrc

        // outer BN + ReLU (in place) -> features for next layer / pooling
        zero(gsum, 2 * (size_t)H2);
        k_bn_stats<<<768, H, 0, stream>>>(xsrc, gsum, gsq, N, H);
        k_bn_finalize<<<1, H, 0, stream>>>(gsum, gsq, bn_g[l], bn_b[l],
                                           scale, shift, N, H);
        k_bn_relu<<<(unsigned)((NH + Z - 1) / Z), Z, 0, stream>>>(xsrc, xsrc,
                                                                  scale, shift,
                                                                  NH, H);
    }

    // ---- global mean pool + final linear ----
    zero(pooled, (size_t)G * H + G);
    k_pool<<<(unsigned)((NH + Z - 1) / Z), Z, 0, stream>>>(xsrc, batch, pooled, cnt, N);
    k_final_linear<<<G, OUT, 0, stream>>>(pooled, cnt, lin_w, lin_b,
                                          (float*)d_out, OUT);
    (void)out_size;
}
